// MultiHeadAttention_47004122087831
// MI455X (gfx1250) — compile-verified, matrix operand order checked by reading, not run on hardware
//
#include <hip/hip_runtime.h>
#include <hip/hip_bf16.h>

// ---------------------------------------------------------------------------
// MI455X (gfx1250) multi-head attention, bf16 WMMA pipeline, fp32 softmax.
// ---------------------------------------------------------------------------

typedef __attribute__((ext_vector_type(16))) __bf16 v16bf;
typedef __attribute__((ext_vector_type(8)))  __bf16 v8bf;
typedef __attribute__((ext_vector_type(8)))  float  v8f;

constexpr int B_  = 4;
constexpr int S_  = 2048;
constexpr int M_  = 1024;
constexpr int H_  = 16;
constexpr int DK_ = 64;
constexpr int TOK = B_ * S_;   // 8192 tokens

#define DEV static __device__ __forceinline__

// A-fragment / B-fragment loader for 16x16x32 bf16 WMMA from a row-major
// matrix. Per ISA layout: lane l holds row (l&15); elements 0..7 are
// K = 8*hi..8*hi+7 and elements 8..15 are K = 16+8*hi..16+8*hi+7 (hi = l>=16).
// Both halves are contiguous 16-byte loads.
DEV v16bf load_frag(const __bf16* p, int hi) {
  union { v16bf v; v8bf h[2]; } u;
  u.h[0] = *reinterpret_cast<const v8bf*>(p + 8 * hi);
  u.h[1] = *reinterpret_cast<const v8bf*>(p + 16 + 8 * hi);
  return u.v;
}

DEV v8f wmma_bf16(v16bf a, v16bf b, v8f c) {
  // (neg_a, A, neg_b, B, c_mod, C, reuse_a, reuse_b)
  return __builtin_amdgcn_wmma_f32_16x16x32_bf16(false, a, false, b,
                                                 (short)0, c, false, false);
}

// ---------------------------------------------------------------------------
// fp32 -> bf16 conversion (vectorized x4)
// ---------------------------------------------------------------------------
__global__ void cvt_f32_bf16_v4(const float* __restrict__ src,
                                __bf16* __restrict__ dst, int n4) {
  int i = blockIdx.x * blockDim.x + threadIdx.x;
  if (i >= n4) return;
  const float4 f = reinterpret_cast<const float4*>(src)[i];
  union { __bf16 b[4]; unsigned long long u; } p;
  p.b[0] = (__bf16)f.x; p.b[1] = (__bf16)f.y;
  p.b[2] = (__bf16)f.z; p.b[3] = (__bf16)f.w;
  reinterpret_cast<unsigned long long*>(dst)[i] = p.u;
}

// ---------------------------------------------------------------------------
// Projection GEMM:  out[m][n] = (sum_k A[m][k] * W[n][k] + bias[n]) * oscale
// A: [TOK][1024] bf16 row-major, W: [1024][1024] bf16 row-major.
// MODE 0: store bf16, concat layout [TOK][1024]           (Q, K projections)
// MODE 1: store bf16 transposed per head: Vt[b][h][d][s]  (V projection)
// MODE 2: store f32, concat layout                         (output projection)
// Block = 8 waves (4 row x 2 col); wave computes a 16x128 strip
// (A-fragment reused 8x); block covers 64 rows x 256 cols.
// ---------------------------------------------------------------------------
template <int MODE>
__global__ void __launch_bounds__(256)
gemm_proj(const __bf16* __restrict__ A, const __bf16* __restrict__ W,
          const float* __restrict__ bias, float* __restrict__ outF,
          __bf16* __restrict__ outB, float oscale) {
  const int tid  = threadIdx.x;
  const int lane = tid & 31;
  const int w    = tid >> 5;
  const int lo   = lane & 15;
  const int hi   = lane >> 4;
  const int r0   = blockIdx.x * 64 + (w >> 1) * 16;   // token rows
  const int n0   = blockIdx.y * 256 + (w & 1) * 128;  // feature cols

  v8f acc[8];
#pragma unroll
  for (int j = 0; j < 8; ++j)
#pragma unroll
    for (int e = 0; e < 8; ++e) acc[j][e] = 0.0f;

  const __bf16* arow = A + (size_t)(r0 + lo) * M_;
#pragma unroll 2
  for (int k0 = 0; k0 < M_; k0 += 32) {
    const v16bf af = load_frag(arow + k0, hi);
#pragma unroll
    for (int j = 0; j < 8; ++j) {
      const __bf16* wrow = W + (size_t)(n0 + j * 16 + lo) * M_ + k0;
      acc[j] = wmma_bf16(af, load_frag(wrow, hi), acc[j]);
    }
  }

#pragma unroll
  for (int j = 0; j < 8; ++j) {
    const int   n  = n0 + j * 16 + lo;
    const float bn = bias[n];
    if constexpr (MODE == 1) {
      // transposed per-head store: Vt[(b*H+h)*DK + d][s], 8 consecutive s
      const int h  = n >> 6, d = n & 63;
      const int tb = r0 + 8 * hi;          // first token of this half
      const int b  = tb >> 11;             // S_ == 2048
      const int sb = tb & (S_ - 1);
      v8bf pk;
#pragma unroll
      for (int r = 0; r < 8; ++r) pk[r] = (__bf16)((acc[j][r] + bn) * oscale);
      *reinterpret_cast<v8bf*>(
          outB + ((size_t)(b * H_ + h) * DK_ + d) * S_ + sb) = pk;
    } else {
#pragma unroll
      for (int r = 0; r < 8; ++r) {
        const int   row = r0 + r + 8 * hi;
        const float v   = (acc[j][r] + bn) * oscale;
        if constexpr (MODE == 0)
          outB[(size_t)row * M_ + n] = (__bf16)v;
        else
          outF[(size_t)row * M_ + n] = v;
      }
    }
  }
}

// ---------------------------------------------------------------------------
// Flash attention: per wave, 16 queries x full key stream, 64 keys/iter.
// scores = (Q' K^T) * mask  with Q' pre-scaled by 1/sqrt(dk);
// online softmax; out += P @ V.  Row-sums computed with WMMAs against an
// all-ones B-fragment (rowsum(P) = P @ ones), avoiding cross-lane shuffles.
// qb/kb: bf16 concat layout [B*S][1024]; vt: bf16 [B*H][DK][S];
// ob: bf16 concat layout [B*S][1024].
// ---------------------------------------------------------------------------
__global__ void __launch_bounds__(256)
attn_kernel(const __bf16* __restrict__ qb, const __bf16* __restrict__ kb,
            const __bf16* __restrict__ vt, const float* __restrict__ mask,
            __bf16* __restrict__ ob) {
  __shared__ __align__(16) __bf16 plds[8][16][64];  // per-wave P staging, 16KB

  const int tid  = threadIdx.x;
  const int lane = tid & 31;
  const int w    = tid >> 5;
  const int lo   = lane & 15;
  const int hi   = lane >> 4;
  const int bh   = blockIdx.y;
  const int b    = bh >> 4;
  const int h    = bh & 15;
  const int q0   = blockIdx.x * 128 + w * 16;  // query strip within (b,h)

  const __bf16* Qb = qb + (size_t)b * S_ * M_ + h * DK_;
  const __bf16* Kb = kb + (size_t)b * S_ * M_ + h * DK_;
  const __bf16* Vt = vt + (size_t)bh * DK_ * S_;
  const float*  Mk = mask + (size_t)b * S_ * S_;

  const __bf16* qrow = Qb + (size_t)(q0 + lo) * M_;
  const v16bf qf0 = load_frag(qrow, hi);        // dims 0..31
  const v16bf qf1 = load_frag(qrow + 32, hi);   // dims 32..63

  v8f zacc;
#pragma unroll
  for (int e = 0; e < 8; ++e) zacc[e] = 0.0f;
  v16bf ones;
#pragma unroll
  for (int e = 0; e < 16; ++e) ones[e] = (__bf16)1.0f;

  v8f o[4];
#pragma unroll
  for (int j = 0; j < 4; ++j) o[j] = zacc;
  float mr[8], lr[8];
#pragma unroll
  for (int r = 0; r < 8; ++r) { mr[r] = -3.0e38f; lr[r] = 0.0f; }

  for (int kt = 0; kt < S_; kt += 64) {
    // scores for 4 key tiles (keys kt .. kt+63)
    v8f s[4];
#pragma unroll
    for (int t = 0; t < 4; ++t) {
      const __bf16* kr = Kb + (size_t)(kt + t * 16 + lo) * M_;
      s[t] = wmma_bf16(qf1, load_frag(kr + 32, hi), zacc);
      s[t] = wmma_bf16(qf0, load_frag(kr, hi), s[t]);
    }

    if (kt + 64 < S_) {  // prefetch next K/V tiles -> global_prefetch_b8
      __builtin_prefetch(Kb + (size_t)(kt + 64 + lo) * M_, 0, 3);
      __builtin_prefetch(Vt + (size_t)lo * S_ + kt + 64, 0, 3);
    }

#pragma unroll
    for (int r = 0; r < 8; ++r) {
      const float* mrow = Mk + (size_t)(q0 + r + 8 * hi) * S_ + kt;
      float a[4];
#pragma unroll
      for (int t = 0; t < 4; ++t) a[t] = s[t][r] * mrow[t * 16 + lo];
      float vmax = fmaxf(fmaxf(a[0], a[1]), fmaxf(a[2], a[3]));
#pragma unroll
      for (int off = 8; off >= 1; off >>= 1)
        vmax = fmaxf(vmax, __shfl_xor(vmax, off, 32));
      const float mnew = fmaxf(mr[r], vmax);
      const float corr = __expf(mr[r] - mnew);
      lr[r] *= corr;   // row-sum added below via ones-WMMA
      mr[r] = mnew;
#pragma unroll
      for (int j = 0; j < 4; ++j) o[j][r] *= corr;
#pragma unroll
      for (int t = 0; t < 4; ++t)
        plds[w][r + 8 * hi][t * 16 + lo] = (__bf16)__expf(a[t] - mnew);
    }
    // wave-private LDS bounce; DS ops are in-order per wave, just wait
    asm volatile("s_wait_dscnt 0" ::: "memory");
    const v16bf pf0 = load_frag(&plds[w][lo][0], hi);   // P keys kt..kt+31
    const v16bf pf1 = load_frag(&plds[w][lo][32], hi);  // P keys kt+32..kt+63

    // rowsum(P) via matrix pipe: every column of (P @ ones) is the row sum
    v8f ssum = wmma_bf16(pf0, ones, zacc);
    ssum     = wmma_bf16(pf1, ones, ssum);
#pragma unroll
    for (int r = 0; r < 8; ++r) lr[r] += ssum[r];

#pragma unroll
    for (int j = 0; j < 4; ++j) {
      const __bf16* vrow = Vt + (size_t)(j * 16 + lo) * S_ + kt;
      o[j] = wmma_bf16(pf0, load_frag(vrow, hi), o[j]);
      o[j] = wmma_bf16(pf1, load_frag(vrow + 32, hi), o[j]);
    }
  }

#pragma unroll
  for (int r = 0; r < 8; ++r) {
    const float inv = __builtin_amdgcn_rcpf(lr[r]);
    const int   row = q0 + r + 8 * hi;
    __bf16* orow = ob + (size_t)(b * S_ + row) * M_ + h * DK_;
#pragma unroll
    for (int j = 0; j < 4; ++j) orow[j * 16 + lo] = (__bf16)(o[j][r] * inv);
  }
}

// ---------------------------------------------------------------------------
extern "C" void kernel_launch(void* const* d_in, const int* in_sizes, int n_in,
                              void* d_out, int out_size, void* d_ws,
                              size_t ws_size, hipStream_t stream) {
  (void)in_sizes; (void)n_in; (void)out_size; (void)ws_size;

  const float* x    = (const float*)d_in[0];
  const float* mask = (const float*)d_in[1];
  const float* Wq   = (const float*)d_in[2];
  const float* bq   = (const float*)d_in[3];
  const float* Wk   = (const float*)d_in[4];
  const float* bk   = (const float*)d_in[5];
  const float* Wv   = (const float*)d_in[6];
  const float* bv   = (const float*)d_in[7];
  const float* Wo   = (const float*)d_in[8];
  const float* bo   = (const float*)d_in[9];

  // Workspace carve-up (all 16B aligned). xb region is reused for the
  // attention output (xb is dead after the V projection).
  char* ws = (char*)d_ws;
  __bf16* xb  = (__bf16*)ws;  ws += (size_t)TOK * M_ * 2;   // 16 MB (also ob)
  __bf16* wqb = (__bf16*)ws;  ws += (size_t)M_ * M_ * 2;    //  2 MB
  __bf16* wkb = (__bf16*)ws;  ws += (size_t)M_ * M_ * 2;
  __bf16* wvb = (__bf16*)ws;  ws += (size_t)M_ * M_ * 2;
  __bf16* wob = (__bf16*)ws;  ws += (size_t)M_ * M_ * 2;
  __bf16* qbuf = (__bf16*)ws; ws += (size_t)TOK * M_ * 2;   // 16 MB
  __bf16* kbuf = (__bf16*)ws; ws += (size_t)TOK * M_ * 2;   // 16 MB
  __bf16* vtb  = (__bf16*)ws; ws += (size_t)TOK * M_ * 2;   // 16 MB (Vt)
  __bf16* obuf = xb;                                        // reuse

  // 1) convert activations + weights to bf16
  {
    const int n4x = TOK * M_ / 4;
    cvt_f32_bf16_v4<<<(n4x + 255) / 256, 256, 0, stream>>>(x, xb, n4x);
    const int n4w = M_ * M_ / 4;
    cvt_f32_bf16_v4<<<(n4w + 255) / 256, 256, 0, stream>>>(Wq, wqb, n4w);
    cvt_f32_bf16_v4<<<(n4w + 255) / 256, 256, 0, stream>>>(Wk, wkb, n4w);
    cvt_f32_bf16_v4<<<(n4w + 255) / 256, 256, 0, stream>>>(Wv, wvb, n4w);
    cvt_f32_bf16_v4<<<(n4w + 255) / 256, 256, 0, stream>>>(Wo, wob, n4w);
  }

  // 2) QKV projections (Q pre-scaled by 1/sqrt(dk); V stored transposed)
  dim3 gb(TOK / 64, M_ / 256);
  gemm_proj<0><<<gb, 256, 0, stream>>>(xb, wqb, bq, nullptr, qbuf, 0.125f);
  gemm_proj<0><<<gb, 256, 0, stream>>>(xb, wkb, bk, nullptr, kbuf, 1.0f);
  gemm_proj<1><<<gb, 256, 0, stream>>>(xb, wvb, bv, nullptr, vtb, 1.0f);

  // 3) flash attention -> concat layout bf16
  dim3 ga(S_ / 128, B_ * H_);
  attn_kernel<<<ga, 256, 0, stream>>>(qbuf, kbuf, vtb, mask, obuf);

  // 4) output projection -> fp32 d_out
  gemm_proj<2><<<gb, 256, 0, stream>>>(obuf, wob, bo, (float*)d_out, nullptr,
                                       1.0f);
}
